// LSTMModel_45088566674125
// MI455X (gfx1250) — matrix operand order repaired and checked by reading, not compiled, not run on hardware
//
#include <hip/hip_runtime.h>
#include <hip/hip_bf16.h>

// LSTM scan for MI455X (gfx1250, wave32).
// B=out_size batches, T steps, D=1, H=50, gates 4H=200.
// Strategy: 16-batch tile per workgroup; per step z[16,200] = h@Wh + x_t*Wx + b
// computed with V_WMMA_F32_16X16X32_F16 (K padded 50->64 => 2 WMMA per N-tile).
// Wh is packed once (prologue kernel) into the exact f16 B-operand lane layout
// and held register-resident across all T steps. h is ping-ponged through LDS
// in the exact f16 A-operand layout (two contiguous 16B rows per half).

#define LSTM_H   50
#define GATES    (4 * LSTM_H)      // 200
#define NTILES   13                // ceil(200/16)
#define NPAD     (NTILES * 16)     // 208
#define KPAD     64                // K padded 50 -> 64 (2 k-blocks of 32)
#define BTILE    16                // batches per workgroup
#define TPB      128               // 4 wave32s

typedef __attribute__((ext_vector_type(16))) _Float16 v16h;
typedef __attribute__((ext_vector_type(8)))  _Float16 v8h;
typedef __attribute__((ext_vector_type(8)))  float    v8f;

// ---------------------------------------------------------------------------
// Prologue: pack Wh[f32, 50x200 row-major] into f16 WMMA B-operand layout.
// Per (ntile, kblock): 32 lanes x 16 halves. VGPR j of lane L holds the pair
// (K = kb*32 + (L/16)*16 + 2j, +1) for column N = nt*16 + (L%16).
// Out-of-range K (>=50) or N (>=200) are zero-filled.
// ---------------------------------------------------------------------------
__global__ void lstm_pack_wh(const float* __restrict__ Wh,
                             _Float16* __restrict__ Bpack) {
  int idx = blockIdx.x * blockDim.x + threadIdx.x;
  const int total = NTILES * 2 * 32 * 16;
  if (idx >= total) return;
  int nt   = idx / (2 * 32 * 16);
  int r    = idx % (2 * 32 * 16);
  int kb   = r / (32 * 16);
  r        = r % (32 * 16);
  int lane = r / 16;
  int e    = r % 16;             // j*2 + p
  int j    = e >> 1;
  int p    = e & 1;
  int k    = kb * 32 + (lane >> 4) * 16 + 2 * j + p;
  int n    = nt * 16 + (lane & 15);
  float v  = (k < LSTM_H && n < GATES) ? Wh[k * GATES + n] : 0.0f;
  Bpack[idx] = (_Float16)v;
}

__device__ __forceinline__ float sigmoidf_(float v) {
  return 1.0f / (1.0f + __expf(-v));
}

// ---------------------------------------------------------------------------
// Main scan kernel: one workgroup = 16 batches, 4 waves split the 13 N-tiles.
// ---------------------------------------------------------------------------
__global__ void __launch_bounds__(TPB)
lstm_scan_kernel(const float* __restrict__ x,      // [B, T]  (D==1)
                 const float* __restrict__ Wx,     // [200]
                 const float* __restrict__ bias,   // [200]
                 const float* __restrict__ Wd,     // [50]
                 const float* __restrict__ bd,     // [1]
                 const _Float16* __restrict__ Bpack,
                 float* __restrict__ out,          // [B]
                 int T) {
  __shared__ float    z_lds[BTILE][NPAD];     // gate pre-activations
  __shared__ float    c_lds[BTILE][LSTM_H];   // cell state (f32, persistent)
  __shared__ float    h32 [BTILE][LSTM_H];    // hidden state (f32, for output)
  __shared__ _Float16 h16 [BTILE][KPAD];      // hidden state in A-operand form
  __shared__ float    x_lds[BTILE];           // x_t staging

  const int tid  = threadIdx.x;
  const int lane = tid & 31;
  const int wave = tid >> 5;
  const int half = lane >> 4;       // 0: lanes 0-15, 1: lanes 16-31
  const int nlo  = lane & 15;
  const int b0   = blockIdx.x * BTILE;

  // --- Register-resident B operands: 4 tile-slots x 2 k-blocks ------------
  v16h Bop[4][2];
  int  ntv[4];
  float wxn[4], bn[4];
#pragma unroll
  for (int s = 0; s < 4; ++s) {
    ntv[s] = wave + 4 * s;                       // 0..15 (13..15 are dummies)
    int nt = ntv[s] < NTILES ? ntv[s] : (NTILES - 1);
#pragma unroll
    for (int kb = 0; kb < 2; ++kb)
      Bop[s][kb] = *(const v16h*)(Bpack + ((nt * 2 + kb) * 32 + lane) * 16);
    int n = ntv[s] * 16 + nlo;
    wxn[s] = (n < GATES) ? Wx[n]   : 0.0f;
    bn[s]  = (n < GATES) ? bias[n] : 0.0f;
  }

  // --- Init: c=0, h=0 (incl. K-padding of h16), stage x_0 ------------------
  for (int i = tid; i < BTILE * LSTM_H; i += TPB)
    c_lds[i / LSTM_H][i % LSTM_H] = 0.0f;
  for (int i = tid; i < BTILE * KPAD; i += TPB)
    h16[i / KPAD][i % KPAD] = (_Float16)0.0f;
  if (tid < BTILE) x_lds[tid] = x[(size_t)(b0 + tid) * T];
  __syncthreads();

  // --- Time scan -----------------------------------------------------------
  for (int t = 0; t < T; ++t) {
    // Build A operands from h16 (exact A layout => 4x ds_load_b128 per lane).
    const int m = nlo;
    v8h a0lo = *(const v8h*)&h16[m][half * 16];       // v0-3: K=0-7 / 16-23
    v8h a0hi = *(const v8h*)&h16[m][half * 16 + 8];   // v4-7: K=8-15 / 24-31
    v8h a1lo = *(const v8h*)&h16[m][32 + half * 16];  // K=32-39 / 48-55
    v8h a1hi = *(const v8h*)&h16[m][40 + half * 16];  // K=40-47 / 56-63 (zeros >=50)
    v16h A0 = __builtin_shufflevector(a0lo, a0hi, 0,1,2,3,4,5,6,7,
                                      8,9,10,11,12,13,14,15);
    v16h A1 = __builtin_shufflevector(a1lo, a1hi, 0,1,2,3,4,5,6,7,
                                      8,9,10,11,12,13,14,15);
    float xv[8];
#pragma unroll
    for (int r = 0; r < 8; ++r) xv[r] = x_lds[half * 8 + r];

#pragma unroll
    for (int s = 0; s < 4; ++s) {
      // Rank-1 + bias term folded into accumulator init (D layout: M=r+8*half).
      v8f acc;
#pragma unroll
      for (int r = 0; r < 8; ++r) acc[r] = xv[r] * wxn[s] + bn[s];
      acc = __builtin_amdgcn_wmma_f32_16x16x32_f16(false, A0, false, Bop[s][0],
                                                   (short)0, acc, false, false);
      acc = __builtin_amdgcn_wmma_f32_16x16x32_f16(false, A1, false, Bop[s][1],
                                                   (short)0, acc, false, false);
      if (ntv[s] < NTILES) {                 // wave-uniform: EXEC full for WMMA
        int col = ntv[s] * 16 + nlo;
#pragma unroll
        for (int r = 0; r < 8; ++r) z_lds[r + 8 * half][col] = acc[r];
      }
    }
    __syncthreads();

    // Gate phase: 800 (b,h) cells over 128 threads, full f32.
    for (int i = tid; i < BTILE * LSTM_H; i += TPB) {
      int b  = i / LSTM_H;
      int hh = i % LSTM_H;
      float ig = sigmoidf_(z_lds[b][hh]);
      float fg = sigmoidf_(z_lds[b][LSTM_H + hh]);
      float gg = tanhf    (z_lds[b][2 * LSTM_H + hh]);
      float og = sigmoidf_(z_lds[b][3 * LSTM_H + hh]);
      float c  = fg * c_lds[b][hh] + ig * gg;
      c_lds[b][hh] = c;
      float h = og * tanhf(c);
      h32[b][hh] = h;
      h16[b][hh] = (_Float16)h;
    }
    if (tid < BTILE && (t + 1) < T)
      x_lds[tid] = x[(size_t)(b0 + tid) * T + (t + 1)];
    __syncthreads();
  }

  // --- Head: out[b] = h_last . Wd + bd ------------------------------------
  if (tid < BTILE) {
    float s = bd[0];
#pragma unroll 5
    for (int hh = 0; hh < LSTM_H; ++hh) s += h32[tid][hh] * Wd[hh];
    out[b0 + tid] = s;
  }
}

// ---------------------------------------------------------------------------
extern "C" void kernel_launch(void* const* d_in, const int* in_sizes, int n_in,
                              void* d_out, int out_size, void* d_ws, size_t ws_size,
                              hipStream_t stream) {
  const float* x    = (const float*)d_in[0];   // [B,T,1]
  const float* Wx   = (const float*)d_in[1];   // [1,200]
  const float* Wh   = (const float*)d_in[2];   // [50,200]
  const float* bias = (const float*)d_in[3];   // [200]
  const float* Wd   = (const float*)d_in[4];   // [50,1]
  const float* bd   = (const float*)d_in[5];   // [1]
  float* out = (float*)d_out;

  const int B = out_size;                      // 1024
  const int T = in_sizes[0] / B;               // 2048

  _Float16* Bpack = (_Float16*)d_ws;           // 26,624 bytes
  const int packN = NTILES * 2 * 32 * 16;
  lstm_pack_wh<<<(packN + 255) / 256, 256, 0, stream>>>(Wh, Bpack);

  lstm_scan_kernel<<<B / BTILE, TPB, 0, stream>>>(
      x, Wx, bias, Wd, bd, Bpack, out, T);
}
